// P4TransformerDA2_78692390797856
// MI455X (gfx1250) — matrix operand order repaired and checked by reading, not compiled, not run on hardware
//
#include <hip/hip_runtime.h>

#define B_    4
#define L_    24
#define N_    512
#define DIM_  1024
#define J_    24
#define M_    64
#define KJM   1536      // J*M
#define FD_   128
#define OUT_  72
#define TILE  128       // tokens per block
#define NBLK  384       // (B*L*N)/TILE

typedef __attribute__((ext_vector_type(16))) __bf16 v16bf;
typedef __attribute__((ext_vector_type(8)))  float  v8f;

union FragB { v16bf v; uint4 u[2]; };
union FragC { v8f  v; float f[8]; };

static __device__ inline v16bf load_frag_2x16B(const __bf16* p0, const __bf16* p1) {
    FragB f;
    f.u[0] = *(const uint4*)p0;
    f.u[1] = *(const uint4*)p1;
    return f.v;
}

// 16 consecutive bf16 (32B) -> one B fragment
static __device__ inline v16bf load_frag_32B(const __bf16* p) {
    return load_frag_2x16B(p, p + 8);
}

static __device__ inline v16bf cvt_frag_f32(const float* lo, const float* hi) {
    FragB f;
#pragma unroll
    for (int i = 0; i < 8; ++i) f.v[i]     = (__bf16)lo[i];
#pragma unroll
    for (int i = 0; i < 8; ++i) f.v[8 + i] = (__bf16)hi[i];
    return f.v;
}

static __device__ inline void atomicMaxF(float* addr, float val) {
    unsigned int* ua  = (unsigned int*)addr;
    unsigned int  old = *ua;
    while (__uint_as_float(old) < val) {
        unsigned int assumed = old;
        old = atomicCAS(ua, assumed, __float_as_uint(val));
        if (old == assumed) break;
    }
}

#define WMMA_BF16(A, Bm, C) \
    __builtin_amdgcn_wmma_f32_16x16x32_bf16(false, (A), false, (Bm), (short)0, (C), false, false)

// ---------------- prepass kernels ----------------

__global__ __launch_bounds__(256) void p4_prep_mem(const float* __restrict__ mem,
                                                   __bf16* __restrict__ memBf,
                                                   __bf16* __restrict__ memT) {
    int i = blockIdx.x * 256 + threadIdx.x;
    if (i < FD_ * KJM) {
        int d = i / KJM, k = i % KJM;
        float v = mem[i];
        memBf[i] = (__bf16)v;                 // [128][1536]
        memT[(long)k * FD_ + d] = (__bf16)v;  // [1536][128]
    }
}

__global__ __launch_bounds__(256) void p4_prep_w(const float* __restrict__ pc_w,
                                                 const float* __restrict__ seg_w,
                                                 __bf16* __restrict__ wcat) {
    int i = blockIdx.x * 256 + threadIdx.x;
    if (i < 160 * DIM_) {
        int r = i / DIM_, k = i % DIM_;
        float v = 0.0f;
        if (r < 128)      v = pc_w[r * DIM_ + k];
        else if (r < 152) v = seg_w[(r - 128) * DIM_ + k];
        wcat[i] = (__bf16)v;                  // [160][1024]: rows 0-127 pc, 128-151 seg, rest 0
    }
}

__global__ __launch_bounds__(256) void p4_init(float* __restrict__ skl) {
    int i = blockIdx.x * 256 + threadIdx.x;
    if (i < B_ * FD_) skl[i] = -3.0e38f;
}

// ---------------- main fused kernel ----------------

__global__ __launch_bounds__(256) void p4_main(
        const float*  __restrict__ feat,
        const float*  __restrict__ seg_b,
        const float*  __restrict__ pc_b,
        const __bf16* __restrict__ memBf,   // [128][1536]
        const __bf16* __restrict__ memT,    // [1536][128]
        const __bf16* __restrict__ wcat,    // [160][1024]
        float* __restrict__ seg_out,        // (B,L,J,N)
        float* __restrict__ skl,            // (B,128), pre-init -inf
        float* __restrict__ lossPart)       // (NBLK,)
{
    __shared__ __align__(16) char smem[131136];
    float*  pcLDS   = (float*)smem;                  // [128][128] f32, 64KB
    float*  segLDS  = (float*)(smem + 65536);        // [128][32]  f32, 16KB
    char*   scrA    = smem + 81920;                  // 48KB overlay region
    __bf16* featBuf = (__bf16*)scrA;                 // phase1: [128][64] bf16
    __bf16* wBuf    = (__bf16*)(scrA + 16384);       // phase1: [160][64] bf16
    float*  logBuf  = (float*)scrA;                  // phase3: [128][64] f32
    __bf16* attnBuf = (__bf16*)(scrA + 32768);       // phase3: [128][64] bf16
    float*  redLDS  = (float*)(smem + 131072);       // [16]

    const int bid = blockIdx.x;
    const int b   = bid / 96;                 // 96 = L*4 tiles per batch
    const int l   = (bid % 96) / 4;
    const int n0  = (bid % 4) * TILE;
    const long tokBase = ((long)(b * L_ + l)) * N_ + n0;
    const float* featT = feat + tokBase * DIM_;

    const int tid  = threadIdx.x;
    const int lane = tid & 31;
    const int wv   = tid >> 5;                // wave id 0..7
    const int lm   = lane & 15;
    const int hl   = lane >> 4;               // lane half

    // ---------------- Phase 1: [feat]x[pc_w | seg_w] ----------------
    v8f acc[10];
#pragma unroll
    for (int c = 0; c < 10; ++c) acc[c] = (v8f){0.f,0.f,0.f,0.f,0.f,0.f,0.f,0.f};

    for (int kc = 0; kc < 16; ++kc) {
        const int k0g = kc * 64;
        // feat chunk -> bf16 LDS (128x64)
        for (int i = tid; i < 128 * 16; i += 256) {
            int t = i >> 4; int kq = (i & 15) * 4;
            float4 v = *(const float4*)(featT + (long)t * DIM_ + k0g + kq);
            __bf16* d = featBuf + t * 64 + kq;
            d[0] = (__bf16)v.x; d[1] = (__bf16)v.y; d[2] = (__bf16)v.z; d[3] = (__bf16)v.w;
        }
        // wcat chunk -> LDS (160x64 bf16) as 16B copies
        for (int i = tid; i < 1280; i += 256) {
            int r = i >> 3; int kq = (i & 7) * 8;
            *(uint4*)(wBuf + r * 64 + kq) =
                *(const uint4*)(wcat + (long)r * DIM_ + k0g + kq);
        }
        if (kc + 1 < 16)  // hint next feat chunk into cache (global_prefetch_b8)
            __builtin_prefetch(featT + (long)(tid >> 1) * DIM_ + (kc + 1) * 64, 0, 1);
        __syncthreads();

#pragma unroll
        for (int ks = 0; ks < 64; ks += 32) {
            const __bf16* ab = featBuf + (16 * wv + lm) * 64 + ks + hl * 16;
            const __bf16* wbase = wBuf + lm * 64 + ks + hl * 16;   // + ct*1024
            // two B fragments live at all times -> partial dscnt waits
            v16bf afrag = load_frag_2x16B(ab, ab + 8);
            v16bf b0 = load_frag_32B(wbase);
            v16bf b1 = load_frag_32B(wbase + 1024);
#pragma unroll
            for (int ct = 0; ct < 10; ct += 2) {
                acc[ct] = WMMA_BF16(afrag, b0, acc[ct]);
                if (ct + 2 < 10) b0 = load_frag_32B(wbase + (ct + 2) * 1024);
                acc[ct + 1] = WMMA_BF16(afrag, b1, acc[ct + 1]);
                if (ct + 3 < 10) b1 = load_frag_32B(wbase + (ct + 3) * 1024);
            }
        }
        __syncthreads();
    }

    // bias + scatter accumulators to LDS
    {
        FragC fc;
#pragma unroll
        for (int ct = 0; ct < 8; ++ct) {
            int col = ct * 16 + lm;
            float bias = pc_b[col];
            fc.v = acc[ct];
#pragma unroll
            for (int r = 0; r < 8; ++r)
                pcLDS[(16 * wv + r + 8 * hl) * 128 + col] = fc.f[r] + bias;
        }
        fc.v = acc[8];
        {
            int j = lm; float bias = seg_b[j];
#pragma unroll
            for (int r = 0; r < 8; ++r)
                segLDS[(16 * wv + r + 8 * hl) * 32 + j] = fc.f[r] + bias;
        }
        fc.v = acc[9];
        if (lm < 8) {
            int j = 16 + lm; float bias = seg_b[j];
#pragma unroll
            for (int r = 0; r < 8; ++r)
                segLDS[(16 * wv + r + 8 * hl) * 32 + j] = fc.f[r] + bias;
        }
    }
    __syncthreads();

    // ---------------- Phase 2: seg softmax over J + seg_out ----------------
    if (tid < 128) {
        float* row = segLDS + tid * 32;
        float mx = row[0];
        for (int j = 1; j < J_; ++j) mx = fmaxf(mx, row[j]);
        float e[J_]; float s = 0.f;
        for (int j = 0; j < J_; ++j) { e[j] = expf(row[j] - mx); s += e[j]; }
        float inv = 1.f / s;
        for (int j = 0; j < J_; ++j) {
            float v = e[j] * inv;
            row[j] = v;
            seg_out[(((long)(b * L_ + l)) * J_ + j) * N_ + n0 + tid] = v;
        }
    }
    __syncthreads();

    // ---------------- Phase 3: forward_mem (flash softmax over k=1536) ----------------
    const float rsqrtD = 0.08838834764831845f;   // 1/sqrt(128)
    float mI = -3.0e38f, lI = 0.f;               // per-token stats (tid<128)

    const float* pcRow = pcLDS + (16 * wv + lm) * 128;

    // PASS 1: online max/sumexp
    for (int kc = 0; kc < 24; ++kc) {
        const int k0 = kc * 64;
        const int j  = kc;                       // joint index (M=64, chunk aligned)
        v8f lacc[4];
#pragma unroll
        for (int t = 0; t < 4; ++t) lacc[t] = (v8f){0.f,0.f,0.f,0.f,0.f,0.f,0.f,0.f};
#pragma unroll
        for (int dsb = 0; dsb < 4; ++dsb) {
            const int d0 = dsb * 32;
            // all 4 k-tile B fragments issued first; A cvt (VALU) hides L2 latency
            const __bf16* mtb = memT + (long)(k0 + lm) * FD_ + d0 + hl * 16;
            v16bf bb0 = load_frag_32B(mtb);
            v16bf bb1 = load_frag_32B(mtb + 16 * FD_);
            v16bf bb2 = load_frag_32B(mtb + 32 * FD_);
            v16bf bb3 = load_frag_32B(mtb + 48 * FD_);
            const float* ar = pcRow + d0 + hl * 16;
            v16bf afrag = cvt_frag_f32(ar, ar + 8);
            lacc[0] = WMMA_BF16(afrag, bb0, lacc[0]);
            lacc[1] = WMMA_BF16(afrag, bb1, lacc[1]);
            lacc[2] = WMMA_BF16(afrag, bb2, lacc[2]);
            lacc[3] = WMMA_BF16(afrag, bb3, lacc[3]);
        }
        float segv[8];
#pragma unroll
        for (int r = 0; r < 8; ++r) segv[r] = segLDS[(16 * wv + r + 8 * hl) * 32 + j];
#pragma unroll
        for (int kt = 0; kt < 4; ++kt) {
            FragC fc; fc.v = lacc[kt];
#pragma unroll
            for (int r = 0; r < 8; ++r) {
                int tok = 16 * wv + r + 8 * hl;
                logBuf[tok * 64 + kt * 16 + lm] = fc.f[r] * rsqrtD * segv[r];
            }
        }
        __syncthreads();
        if (tid < 128) {
            const float* rowp = logBuf + tid * 64;
            float mx = rowp[0];
            for (int kk = 1; kk < 64; ++kk) mx = fmaxf(mx, rowp[kk]);
            float mN = fmaxf(mI, mx);
            float ssum = 0.f;
            for (int kk = 0; kk < 64; ++kk) ssum += expf(rowp[kk] - mN);
            lI = lI * expf(mI - mN) + ssum;
            mI = mN;
        }
        __syncthreads();
    }

    // PASS 2: attn -> rec = attn^T x mem^T
    v8f racc[8];
#pragma unroll
    for (int t = 0; t < 8; ++t) racc[t] = (v8f){0.f,0.f,0.f,0.f,0.f,0.f,0.f,0.f};

    for (int kc = 0; kc < 24; ++kc) {
        const int k0 = kc * 64;
        const int j  = kc;
        v8f lacc[4];
#pragma unroll
        for (int t = 0; t < 4; ++t) lacc[t] = (v8f){0.f,0.f,0.f,0.f,0.f,0.f,0.f,0.f};
#pragma unroll
        for (int dsb = 0; dsb < 4; ++dsb) {
            const int d0 = dsb * 32;
            const __bf16* mtb = memT + (long)(k0 + lm) * FD_ + d0 + hl * 16;
            v16bf bb0 = load_frag_32B(mtb);
            v16bf bb1 = load_frag_32B(mtb + 16 * FD_);
            v16bf bb2 = load_frag_32B(mtb + 32 * FD_);
            v16bf bb3 = load_frag_32B(mtb + 48 * FD_);
            const float* ar = pcRow + d0 + hl * 16;
            v16bf afrag = cvt_frag_f32(ar, ar + 8);
            lacc[0] = WMMA_BF16(afrag, bb0, lacc[0]);
            lacc[1] = WMMA_BF16(afrag, bb1, lacc[1]);
            lacc[2] = WMMA_BF16(afrag, bb2, lacc[2]);
            lacc[3] = WMMA_BF16(afrag, bb3, lacc[3]);
        }
        float segv[8];
#pragma unroll
        for (int r = 0; r < 8; ++r) segv[r] = segLDS[(16 * wv + r + 8 * hl) * 32 + j];
#pragma unroll
        for (int kt = 0; kt < 4; ++kt) {
            FragC fc; fc.v = lacc[kt];
#pragma unroll
            for (int r = 0; r < 8; ++r) {
                int tok = 16 * wv + r + 8 * hl;
                logBuf[tok * 64 + kt * 16 + lm] = fc.f[r] * rsqrtD * segv[r];
            }
        }
        __syncthreads();
        if (tid < 128) {
            const float* rowp = logBuf + tid * 64;
            __bf16* arow = attnBuf + tid * 64;
            float inv = 1.f / lI;
            for (int kk = 0; kk < 64; ++kk)
                arow[kk] = (__bf16)(expf(rowp[kk] - mI) * inv);
        }
        __syncthreads();
#pragma unroll
        for (int ks = 0; ks < 64; ks += 32) {
            const __bf16* ab = attnBuf + (16 * wv + lm) * 64 + ks + hl * 16;
            const __bf16* mb = memBf + (long)lm * KJM + k0 + ks + hl * 16;  // + dt*16*KJM
            // 4-deep rotation over 8 d-tiles: 4 global fragments always in flight
            v16bf c0 = load_frag_32B(mb);
            v16bf c1 = load_frag_32B(mb + (long)16 * KJM);
            v16bf c2 = load_frag_32B(mb + (long)32 * KJM);
            v16bf c3 = load_frag_32B(mb + (long)48 * KJM);
            v16bf afrag = load_frag_2x16B(ab, ab + 8);
            racc[0] = WMMA_BF16(afrag, c0, racc[0]);
            c0 = load_frag_32B(mb + (long)64 * KJM);
            racc[1] = WMMA_BF16(afrag, c1, racc[1]);
            c1 = load_frag_32B(mb + (long)80 * KJM);
            racc[2] = WMMA_BF16(afrag, c2, racc[2]);
            c2 = load_frag_32B(mb + (long)96 * KJM);
            racc[3] = WMMA_BF16(afrag, c3, racc[3]);
            c3 = load_frag_32B(mb + (long)112 * KJM);
            racc[4] = WMMA_BF16(afrag, c0, racc[4]);
            racc[5] = WMMA_BF16(afrag, c1, racc[5]);
            racc[6] = WMMA_BF16(afrag, c2, racc[6]);
            racc[7] = WMMA_BF16(afrag, c3, racc[7]);
        }
        // logBuf/attnBuf rewrite for next chunk is protected by the next
        // iteration's post-store barrier (all waves have left rec-GEMM by then)
    }
    __syncthreads();

    // ---------------- Epilogue: loss partial + per-batch column max ----------------
    float lsum = 0.f;
    {
        FragC fc;
#pragma unroll
        for (int dt = 0; dt < 8; ++dt) {
            fc.v = racc[dt];
            int col = dt * 16 + lm;
#pragma unroll
            for (int r = 0; r < 8; ++r) {
                int tok = 16 * wv + r + 8 * hl;
                float d = fc.f[r] - pcLDS[tok * 128 + col];
                lsum += d * d;
            }
            float mx = fc.f[0];
#pragma unroll
            for (int r = 1; r < 8; ++r) mx = fmaxf(mx, fc.f[r]);
            mx = fmaxf(mx, __shfl_xor(mx, 16, 32));
            if (hl == 0) logBuf[wv * 128 + dt * 16 + lm] = mx;   // reuse as [8][128]
        }
    }
    for (int o = 16; o > 0; o >>= 1) lsum += __shfl_down(lsum, o, 32);
    if (lane == 0) redLDS[wv] = lsum;
    __syncthreads();
    if (tid == 0) {
        float s = 0.f;
        for (int i = 0; i < 8; ++i) s += redLDS[i];
        lossPart[bid] = s;
    }
    if (tid < 128) {
        float mx = logBuf[tid];
        for (int ww = 1; ww < 8; ++ww) mx = fmaxf(mx, logBuf[ww * 128 + tid]);
        atomicMaxF(skl + b * FD_ + tid, mx);
    }
}

// ---------------- final head kernel ----------------

__global__ __launch_bounds__(128) void p4_final(
        const float* __restrict__ ln_g, const float* __restrict__ ln_b,
        const float* __restrict__ w1,   const float* __restrict__ b1,
        const float* __restrict__ w2,   const float* __restrict__ b2,
        const float* __restrict__ skl,  const float* __restrict__ lossPart,
        float* __restrict__ out)
{
    __shared__ float sh[FD_];
    __shared__ float sh2[FD_ / 2];
    __shared__ float stats[2];
    const int t = threadIdx.x;

    if (t == 0) {
        float s = 0.f;
        for (int i = 0; i < NBLK; ++i) s += lossPart[i];
        out[288 + (long)B_ * L_ * J_ * N_] = s / (float)((long)B_ * (L_ * N_) * FD_);
    }
    for (int b = 0; b < B_; ++b) {
        float v = skl[b * FD_ + t];
        sh[t] = v;
        __syncthreads();
        if (t == 0) {
            float mu = 0.f;
            for (int i = 0; i < FD_; ++i) mu += sh[i];
            mu /= (float)FD_;
            float va = 0.f;
            for (int i = 0; i < FD_; ++i) { float d = sh[i] - mu; va += d * d; }
            va /= (float)FD_;
            stats[0] = mu; stats[1] = rsqrtf(va + 1e-5f);
        }
        __syncthreads();
        float h = (v - stats[0]) * stats[1] * ln_g[t] + ln_b[t];
        sh[t] = h;
        __syncthreads();
        if (t < FD_ / 2) {
            float a = b1[t];
            for (int d = 0; d < FD_; ++d) a += sh[d] * w1[t * FD_ + d];
            sh2[t] = 0.5f * a * (1.f + erff(a * 0.7071067811865476f));
        }
        __syncthreads();
        if (t < OUT_) {
            float y = b2[t];
            for (int jj = 0; jj < FD_ / 2; ++jj) y += sh2[jj] * w2[t * (FD_ / 2) + jj];
            out[b * OUT_ + t] = y;
        }
        __syncthreads();
    }
}

// ---------------- launcher ----------------

extern "C" void kernel_launch(void* const* d_in, const int* in_sizes, int n_in,
                              void* d_out, int out_size, void* d_ws, size_t ws_size,
                              hipStream_t stream) {
    const float* feat  = (const float*)d_in[0];
    const float* seg_w = (const float*)d_in[1];
    const float* seg_b = (const float*)d_in[2];
    const float* pc_w  = (const float*)d_in[3];
    const float* pc_b  = (const float*)d_in[4];
    const float* mem   = (const float*)d_in[5];
    const float* ln_g  = (const float*)d_in[6];
    const float* ln_b  = (const float*)d_in[7];
    const float* w1    = (const float*)d_in[8];
    const float* b1    = (const float*)d_in[9];
    const float* w2    = (const float*)d_in[10];
    const float* b2    = (const float*)d_in[11];
    float* out = (float*)d_out;

    char* ws = (char*)d_ws;
    __bf16* memBf = (__bf16*)(ws + 0);         // 384KB
    __bf16* memT  = (__bf16*)(ws + 393216);    // 384KB
    __bf16* wcat  = (__bf16*)(ws + 786432);    // 320KB
    float*  skl   = (float*)(ws + 1114112);    // 2KB
    float*  lossP = (float*)(ws + 1116160);    // 1.5KB

    hipLaunchKernelGGL(p4_prep_mem, dim3(768), dim3(256), 0, stream, mem, memBf, memT);
    hipLaunchKernelGGL(p4_prep_w,  dim3(640), dim3(256), 0, stream, pc_w, seg_w, wcat);
    hipLaunchKernelGGL(p4_init,    dim3(2),   dim3(256), 0, stream, skl);
    hipLaunchKernelGGL(p4_main,    dim3(NBLK), dim3(256), 0, stream,
                       feat, seg_b, pc_b, memBf, memT, wcat,
                       out + 288, skl, lossP);
    hipLaunchKernelGGL(p4_final,   dim3(1),   dim3(128), 0, stream,
                       ln_g, ln_b, w1, b1, w2, b2, skl, lossP, out);
}
